// MultiHeadAttention_2336462209665
// MI455X (gfx1250) — matrix operand order repaired and checked by reading, not compiled, not run on hardware
//
#include <hip/hip_runtime.h>

#define EMBED 1024
#define HEADS 16
#define HD    64
#define BATCH 4
#define SEQ   2048
#define MROWS (BATCH * SEQ)   // 8192

typedef __attribute__((ext_vector_type(16))) __bf16 v16bf;
typedef __attribute__((ext_vector_type(8)))  float  v8f;
typedef __attribute__((ext_vector_type(4)))  unsigned int v4u;
typedef __attribute__((ext_vector_type(8)))  int v8i;
typedef __attribute__((ext_vector_type(4)))  int v4i;

#if defined(__has_builtin)
#if __has_builtin(__builtin_amdgcn_tensor_load_to_lds)
#define HAVE_TDM 1
#endif
#endif
#ifndef HAVE_TDM
#define HAVE_TDM 0
#endif

struct U8x32 { uint4 lo, hi; };

static __device__ __forceinline__ v16bf pack_ab(uint4 lo, uint4 hi) {
    U8x32 u; u.lo = lo; u.hi = hi;
    return __builtin_bit_cast(v16bf, u);
}

static __device__ __forceinline__ unsigned short f2bf(float f) {
    unsigned int u = __float_as_uint(f);
    u += 0x7FFFu + ((u >> 16) & 1u);   // round-to-nearest-even
    return (unsigned short)(u >> 16);
}

static __device__ __forceinline__ v8f wmma_bf16(v16bf a, v16bf b, v8f c) {
    return __builtin_amdgcn_wmma_f32_16x16x32_bf16(false, a, false, b,
                                                   (short)0, c, false, false);
}

// 16-lane row reductions in pure VALU via DPP row rotates (no LDS traffic).
static __device__ __forceinline__ float row_max16(float v) {
    v = fmaxf(v, __int_as_float(__builtin_amdgcn_update_dpp(0, __float_as_int(v), 0x121, 0xF, 0xF, true)));
    v = fmaxf(v, __int_as_float(__builtin_amdgcn_update_dpp(0, __float_as_int(v), 0x122, 0xF, 0xF, true)));
    v = fmaxf(v, __int_as_float(__builtin_amdgcn_update_dpp(0, __float_as_int(v), 0x124, 0xF, 0xF, true)));
    v = fmaxf(v, __int_as_float(__builtin_amdgcn_update_dpp(0, __float_as_int(v), 0x128, 0xF, 0xF, true)));
    return v;
}
static __device__ __forceinline__ float row_sum16(float v) {
    v += __int_as_float(__builtin_amdgcn_update_dpp(0, __float_as_int(v), 0x121, 0xF, 0xF, true));
    v += __int_as_float(__builtin_amdgcn_update_dpp(0, __float_as_int(v), 0x122, 0xF, 0xF, true));
    v += __int_as_float(__builtin_amdgcn_update_dpp(0, __float_as_int(v), 0x124, 0xF, 0xF, true));
    v += __int_as_float(__builtin_amdgcn_update_dpp(0, __float_as_int(v), 0x128, 0xF, 0xF, true));
    return v;
}

static __device__ __forceinline__ unsigned int lds_off(const void* p) {
    return (unsigned int)(unsigned long long)(uintptr_t)p;
}

// Issue a TDM 2D tile load: tensor[dim1][dim0] (row stride = stride0 elems,
// 2-byte elements) -> LDS, tile = tile_d1 rows x tile_d0 elems, packed.
static __device__ __forceinline__ void tdm_load_2d(
    unsigned int lds_addr, const void* gptr,
    unsigned int tensor_d0, unsigned int tensor_d1,
    unsigned int tile_d0, unsigned int tile_d1,
    unsigned long long stride0)
{
#if HAVE_TDM
    unsigned long long ga = (unsigned long long)(uintptr_t)gptr;
    v4u g0;
    g0[0] = 1u;                                            // count=1 (valid)
    g0[1] = lds_addr;                                      // lds_addr bytes
    g0[2] = (unsigned int)ga;                              // global_addr[31:0]
    g0[3] = (unsigned int)((ga >> 32) & 0x1FFFFFFu) | (2u << 30);  // [56:32] | type=2
    v8i g1;
    g1[0] = (int)(1u << 16);                               // data_size=1 (2B)
    g1[1] = (int)((tensor_d0 & 0xFFFFu) << 16);            // tensor_dim0 lo
    g1[2] = (int)((tensor_d0 >> 16) | ((tensor_d1 & 0xFFFFu) << 16));
    g1[3] = (int)((tensor_d1 >> 16) | (tile_d0 << 16));
    g1[4] = (int)(tile_d1 & 0xFFFFu);                      // tile_dim1, tile_dim2=0
    g1[5] = (int)(stride0 & 0xFFFFFFFFull);                // tensor_dim0_stride lo
    g1[6] = (int)((stride0 >> 32) & 0xFFFFull);            // stride hi, dim1_stride=0
    g1[7] = 0;
    v4i gz = {0, 0, 0, 0};
#if defined(__clang_major__) && __clang_major__ >= 23
    v8i gz8 = {0, 0, 0, 0, 0, 0, 0, 0};
    __builtin_amdgcn_tensor_load_to_lds(g0, g1, gz, gz, gz8, 0);
#else
    __builtin_amdgcn_tensor_load_to_lds(g0, g1, gz, gz, 0);
#endif
#else
    (void)lds_addr; (void)gptr; (void)tensor_d0; (void)tensor_d1;
    (void)tile_d0; (void)tile_d1; (void)stride0;
#endif
}

// ---------------------------------------------------------------------------
// fp32 -> bf16 elementwise conversion
// ---------------------------------------------------------------------------
__global__ void f32_to_bf16_kernel(const float* __restrict__ src,
                                   unsigned short* __restrict__ dst, int n) {
    for (int i = blockIdx.x * blockDim.x + threadIdx.x; i < n;
         i += gridDim.x * blockDim.x)
        dst[i] = f2bf(src[i]);
}

// ---------------------------------------------------------------------------
// WMMA bf16 GEMM: C[M,N] = A[M,K] * W[N,K]^T + bias[N]
// Block tile 128M x 128N, 8 waves as 4(row)x2(col); wave tile 32x64.
// A/B tiles staged per k-step into double-buffered LDS via TDM.
// MODE 0: QKV epilogue -> scatter Q (x0.125), K, V^T as bf16
// MODE 1: out-proj epilogue -> fp32 store to d_out
// ---------------------------------------------------------------------------
template <int MODE>
__global__ __launch_bounds__(256) void gemm_wmma_bf16(
    const unsigned short* __restrict__ A,   // [M,K] bf16 row-major
    const unsigned short* __restrict__ W,   // [N,K] bf16 row-major
    const float* __restrict__ bias,         // [N]
    unsigned short* __restrict__ qo,        // MODE 0: [B,H,S,HD]
    unsigned short* __restrict__ ko,        // MODE 0: [B,H,S,HD]
    unsigned short* __restrict__ vto,       // MODE 0: [B,H,HD,S]
    float* __restrict__ fo,                 // MODE 1: [M,N]
    int M, int N_total, int K)
{
    __shared__ __align__(128) unsigned short abuf[2][128 * 32];  // 8 KB each
    __shared__ __align__(128) unsigned short bbuf[2][128 * 32];

    const int tid  = threadIdx.x;
    const int lane = tid & 31;
    const int wave = tid >> 5;
    const int lm   = lane & 15;
    const int hi   = lane >> 4;
    const int wr   = wave & 3;            // 4 row groups of 32
    const int wc   = wave >> 2;           // 2 col groups of 64
    const int blk_row = blockIdx.x * 128;
    const int blk_col = blockIdx.y * 128;

    const int a_koff = hi ? 8 : 0;
    const int b_koff = hi ? 8 : 0;

    v8f acc[2][4];
#pragma unroll
    for (int rt = 0; rt < 2; ++rt)
#pragma unroll
        for (int t = 0; t < 4; ++t) acc[rt][t] = v8f{};

    auto stage = [&](int buf, int k0) {
#if HAVE_TDM
        if (wave == 0) {
            tdm_load_2d(lds_off(&abuf[buf][0]),
                        A + (size_t)blk_row * K + k0,
                        K, M, 32, 128, K);
            tdm_load_2d(lds_off(&bbuf[buf][0]),
                        W + (size_t)blk_col * K + k0,
                        K, N_total, 32, 128, K);
        }
#else
#pragma unroll
        for (int rep = 0; rep < 2; ++rep) {
            const int j  = tid + rep * 256;          // 512 uint4 per tile
            const int r  = j >> 2;
            const int c8 = (j & 3) * 8;
            *(uint4*)&abuf[buf][r * 32 + c8] =
                *(const uint4*)(A + (size_t)(blk_row + r) * K + k0 + c8);
            *(uint4*)&bbuf[buf][r * 32 + c8] =
                *(const uint4*)(W + (size_t)(blk_col + r) * K + k0 + c8);
        }
#endif
    };

    stage(0, 0);

    for (int k0 = 0; k0 < K; k0 += 32) {
        const int cur = (k0 >> 5) & 1;
#if HAVE_TDM
        if (wave == 0) __builtin_amdgcn_s_wait_tensorcnt(0);
#endif
        __syncthreads();
        if (k0 + 32 < K) stage(cur ^ 1, k0 + 32);

        const unsigned short* ap0 = &abuf[cur][(wr * 32 + lm) * 32 + a_koff];
        const unsigned short* ap1 = &abuf[cur][(wr * 32 + 16 + lm) * 32 + a_koff];
        v16bf af0 = pack_ab(*(const uint4*)(ap0), *(const uint4*)(ap0 + 16));
        v16bf af1 = pack_ab(*(const uint4*)(ap1), *(const uint4*)(ap1 + 16));
#pragma unroll
        for (int t = 0; t < 4; ++t) {
            const unsigned short* bp =
                &bbuf[cur][(wc * 64 + t * 16 + lm) * 32 + b_koff];
            v16bf bf = pack_ab(*(const uint4*)(bp), *(const uint4*)(bp + 16));
            acc[0][t] = wmma_bf16(af0, bf, acc[0][t]);
            acc[1][t] = wmma_bf16(af1, bf, acc[1][t]);
        }
        __syncthreads();
    }

#pragma unroll
    for (int rt = 0; rt < 2; ++rt) {
        const int row_base = blk_row + wr * 32 + rt * 16;
#pragma unroll
        for (int t = 0; t < 4; ++t) {
            const int col = blk_col + wc * 64 + t * 16 + lm;
            const float bv = bias[col];
            if (MODE == 0) {
                const int h  = col / (3 * HD);
                const int rr = col % (3 * HD);
#pragma unroll
                for (int i = 0; i < 8; ++i) {
                    const int r = row_base + i + hi * 8;   // flat B*S row
                    const int b = r >> 11;                 // / SEQ
                    const int s = r & (SEQ - 1);
                    const float val = acc[rt][t][i] + bv;
                    const size_t bh = (size_t)(b * HEADS + h);
                    if (rr < HD) {
                        qo[(bh * SEQ + s) * HD + rr] = f2bf(val * 0.125f);
                    } else if (rr < 2 * HD) {
                        ko[(bh * SEQ + s) * HD + (rr - HD)] = f2bf(val);
                    } else {
                        vto[(bh * HD + (rr - 2 * HD)) * SEQ + s] = f2bf(val);
                    }
                }
            } else {
#pragma unroll
                for (int i = 0; i < 8; ++i) {
                    const int r = row_base + i + hi * 8;
                    fo[(size_t)r * N_total + col] = acc[rt][t][i] + bv;
                }
            }
        }
    }
}

// ---------------------------------------------------------------------------
// Flash attention, one (b,h) per blockIdx.y, 128 query rows per block.
// 64-key blocks: K/V tiles double-buffered in LDS via TDM, shared by 8 waves.
// 16 WMMAs per key block per wave; DPP-based softmax reductions.
// ---------------------------------------------------------------------------
__global__ __launch_bounds__(256) void attn_wmma_bf16(
    const unsigned short* __restrict__ Q,    // [B*H, S, HD] bf16
    const unsigned short* __restrict__ Km,   // [B*H, S, HD] bf16
    const unsigned short* __restrict__ Vt,   // [B*H, HD, S] bf16
    unsigned short* __restrict__ ctx)        // [B, S, E]  bf16
{
    __shared__ __align__(128) unsigned short kbuf[2][64 * HD];  // 64 keys x 64 dims
    __shared__ __align__(128) unsigned short vbuf[2][HD * 64];  // 64 dims x 64 keys
    __shared__ __align__(16)  unsigned short plds[8][16 * 64];

    const int tid   = threadIdx.x;
    const int lane  = tid & 31;
    const int wave  = tid >> 5;
    const int lm    = lane & 15;
    const int hi    = lane >> 4;
    const int bh    = blockIdx.y;
    const int qbase = blockIdx.x * 128 + wave * 16;

    const unsigned short* Qb = Q  + (size_t)bh * SEQ * HD;
    const unsigned short* Kb = Km + (size_t)bh * SEQ * HD;
    const unsigned short* Vb = Vt + (size_t)bh * HD * SEQ;

    const int a_koff = hi ? 8 : 0;
    const int b_koff = hi ? 8 : 0;

    // resident Q A-fragments covering head-dim k = 0..63
    const unsigned short* qp = Qb + (size_t)(qbase + lm) * HD;
    v16bf qa0 = pack_ab(*(const uint4*)(qp + a_koff),
                        *(const uint4*)(qp + a_koff + 16));
    v16bf qa1 = pack_ab(*(const uint4*)(qp + 32 + a_koff),
                        *(const uint4*)(qp + 32 + a_koff + 16));

    float mrun[8], lrun[8];
#pragma unroll
    for (int i = 0; i < 8; ++i) { mrun[i] = -1e30f; lrun[i] = 0.0f; }
    v8f o[4] = {v8f{}, v8f{}, v8f{}, v8f{}};

    auto stage = [&](int buf, int kb2) {
#if HAVE_TDM
        if (wave == 0) {
            // K tile: 64 contiguous rows of 64 elems
            tdm_load_2d(lds_off(&kbuf[buf][0]), Kb + (size_t)kb2 * HD,
                        HD, SEQ, HD, 64, HD);
            // V tile: 64 rows x 64 cols from Vt[0..63][kb2..kb2+63]
            tdm_load_2d(lds_off(&vbuf[buf][0]), Vb + kb2,
                        SEQ, HD, 64, HD, SEQ);
        }
#else
#pragma unroll
        for (int rep = 0; rep < 2; ++rep) {
            const int j = tid + rep * 256;               // 512 uint4 per tile
            ((uint4*)&kbuf[buf][0])[j] =
                ((const uint4*)(Kb + (size_t)kb2 * HD))[j];
            const int r  = j >> 3;
            const int c8 = (j & 7) * 8;
            *(uint4*)&vbuf[buf][r * 64 + c8] =
                *(const uint4*)(Vb + (size_t)r * SEQ + kb2 + c8);
        }
#endif
    };

    stage(0, 0);

    for (int kb = 0; kb < SEQ; kb += 64) {
        const int cur = (kb >> 6) & 1;
#if HAVE_TDM
        if (wave == 0) __builtin_amdgcn_s_wait_tensorcnt(0);
#endif
        __syncthreads();                       // current buffer ready for all
        if (kb + 64 < SEQ) stage(cur ^ 1, kb + 64);

        // ---- scores: four 16x16 tiles over keys kb..kb+63 (8 WMMAs, LDS src)
        v8f sf[4];
#pragma unroll
        for (int j = 0; j < 4; ++j) {
            v8f s = {};
            const unsigned short* kp = &kbuf[cur][(j * 16 + lm) * HD + b_koff];
            v16bf b0 = pack_ab(*(const uint4*)(kp),
                               *(const uint4*)(kp + 16));
            s = wmma_bf16(qa0, b0, s);
            v16bf b1 = pack_ab(*(const uint4*)(kp + 32),
                               *(const uint4*)(kp + 48));
            s = wmma_bf16(qa1, b1, s);
            sf[j] = s;
        }

        // ---- online softmax over the 16x64 score block (DPP reductions)
#pragma unroll
        for (int i = 0; i < 8; ++i) {
            float t = fmaxf(fmaxf(sf[0][i], sf[1][i]),
                            fmaxf(sf[2][i], sf[3][i]));
            t = row_max16(t);
            const float mnew  = fmaxf(mrun[i], t);
            const float scale = __expf(mrun[i] - mnew);
            float p[4];
            float psum = 0.0f;
#pragma unroll
            for (int j = 0; j < 4; ++j) {
                p[j] = __expf(sf[j][i] - mnew);
                psum += p[j];
            }
            const float rs = row_sum16(psum);
            lrun[i] = lrun[i] * scale + rs;
            mrun[i] = mnew;
#pragma unroll
            for (int t4 = 0; t4 < 4; ++t4) o[t4][i] *= scale;
            const int row = i + hi * 8;
#pragma unroll
            for (int j = 0; j < 4; ++j)
                plds[wave][row * 64 + j * 16 + lm] = f2bf(p[j]);
        }

        // ---- re-layout P: C-fragment (via per-wave LDS) -> two A-fragments
        const unsigned short* pp0 = &plds[wave][lm * 64 + a_koff];
        const unsigned short* pp1 = &plds[wave][lm * 64 + 32 + a_koff];
        v16bf pa0 = pack_ab(*(const uint4*)(pp0), *(const uint4*)(pp0 + 16));
        v16bf pa1 = pack_ab(*(const uint4*)(pp1), *(const uint4*)(pp1 + 16));

        // ---- O += P x V  (8 WMMAs across head dim, LDS src)
#pragma unroll
        for (int t4 = 0; t4 < 4; ++t4) {
            const unsigned short* vp0 = &vbuf[cur][(t4 * 16 + lm) * 64 + b_koff];
            v16bf vb0 = pack_ab(*(const uint4*)(vp0), *(const uint4*)(vp0 + 16));
            o[t4] = wmma_bf16(pa0, vb0, o[t4]);
            const unsigned short* vp1 =
                &vbuf[cur][(t4 * 16 + lm) * 64 + 32 + b_koff];
            v16bf vb1 = pack_ab(*(const uint4*)(vp1), *(const uint4*)(vp1 + 16));
            o[t4] = wmma_bf16(pa1, vb1, o[t4]);
        }
        __syncthreads();                       // all reads done before reuse
    }

    // ---- finalize: O /= l, store ctx [B,S,E] bf16
    const int b = bh >> 4;      // / HEADS
    const int h = bh & 15;
#pragma unroll
    for (int i = 0; i < 8; ++i) {
        const float inv = 1.0f / lrun[i];
        const int s = qbase + i + hi * 8;
        const size_t rowoff = ((size_t)b * SEQ + s) * EMBED + h * HD;
#pragma unroll
        for (int t4 = 0; t4 < 4; ++t4)
            ctx[rowoff + t4 * 16 + lm] = f2bf(o[t4][i] * inv);
    }
}

// ---------------------------------------------------------------------------
extern "C" void kernel_launch(void* const* d_in, const int* in_sizes, int n_in,
                              void* d_out, int out_size, void* d_ws, size_t ws_size,
                              hipStream_t stream) {
    (void)in_sizes; (void)n_in; (void)out_size; (void)ws_size;

    const float* x     = (const float*)d_in[0];
    const float* qkv_w = (const float*)d_in[1];
    const float* qkv_b = (const float*)d_in[2];
    const float* out_w = (const float*)d_in[3];
    const float* out_b = (const float*)d_in[4];
    float* out = (float*)d_out;

    char* ws = (char*)d_ws;
    size_t off = 0;
    auto alloc = [&](size_t bytes) -> char* {
        char* p = ws + off;
        off += (bytes + 255) & ~(size_t)255;
        return p;
    };
    unsigned short* xb   = (unsigned short*)alloc((size_t)MROWS * EMBED * 2);
    unsigned short* wqkv = (unsigned short*)alloc((size_t)3 * EMBED * EMBED * 2);
    unsigned short* wout = (unsigned short*)alloc((size_t)EMBED * EMBED * 2);
    unsigned short* qb   = (unsigned short*)alloc((size_t)MROWS * EMBED * 2);
    unsigned short* kb   = (unsigned short*)alloc((size_t)MROWS * EMBED * 2);
    unsigned short* vtb  = (unsigned short*)alloc((size_t)MROWS * EMBED * 2);
    unsigned short* ctx  = (unsigned short*)alloc((size_t)MROWS * EMBED * 2);

    f32_to_bf16_kernel<<<1024, 256, 0, stream>>>(x, xb, MROWS * EMBED);
    f32_to_bf16_kernel<<<1024, 256, 0, stream>>>(qkv_w, wqkv, 3 * EMBED * EMBED);
    f32_to_bf16_kernel<<<512,  256, 0, stream>>>(out_w, wout, EMBED * EMBED);

    // QKV projection: M=8192, N=3072, K=1024
    gemm_wmma_bf16<0><<<dim3(MROWS / 128, (3 * EMBED) / 128), 256, 0, stream>>>(
        xb, wqkv, qkv_b, qb, kb, vtb, nullptr, MROWS, 3 * EMBED, EMBED);

    // attention
    attn_wmma_bf16<<<dim3(SEQ / 128, BATCH * HEADS), 256, 0, stream>>>(
        qb, kb, vtb, ctx);

    // output projection: M=8192, N=1024, K=1024
    gemm_wmma_bf16<1><<<dim3(MROWS / 128, EMBED / 128), 256, 0, stream>>>(
        ctx, wout, out_b, nullptr, nullptr, nullptr, out, MROWS, EMBED, EMBED);
}
